// Attention_28630251995310
// MI455X (gfx1250) — matrix operand order repaired and checked by reading, not compiled, not run on hardware
//
#include <hip/hip_runtime.h>
#include <type_traits>

typedef __bf16 bf16_t;
typedef __attribute__((ext_vector_type(16))) __bf16 v16bf;
typedef __attribute__((ext_vector_type(8)))  __bf16 v8bf;
typedef __attribute__((ext_vector_type(8)))  float   v8f;
typedef __attribute__((ext_vector_type(4)))  int     v4i_t;

#define NH     32
#define NKVH   8
#define HD     128
#define GROUP  4           // NH / NKVH
#define SEQQ   1024
#define CTXN   3072
#define LTOT   (CTXN + SEQQ)
#define NSLOTS 8192
// 1/sqrt(128) * log2(e): softmax computed in exp2 domain (invariant rescale)
#define QSCALE (0.08838834764831845f * 1.4426950408889634f)
#define BM     16          // query rows per wave tile
#define CH     64          // keys per chunk
#define NF     (CH / 16)   // 4 score sub-tiles per chunk
#define KPAD   (HD + 8)    // 136 halfs = 272B rows (16B-aligned, conflict-free)
#define VPAD   (CH + 8)    // 72 halfs = 144B rows
#define PPAD   (CH + 8)    // 72 halfs
#define NEGINF (-1e30f)

// LDS overlay offsets (bytes)
#define KSB_BYTES  (2 * CH * KPAD * 2)          // 34816 (double-buffered K)
#define OFF_VT     KSB_BYTES                    // 34816
#define VT_BYTES   (HD * VPAD * 2)              // 18432 (also Q staging area)
#define OFF_PS     (OFF_VT + VT_BYTES)          // 53248
#define PS_BYTES   (GROUP * BM * PPAD * 2)      // 9216
#define SMEM_BYTES (OFF_PS + PS_BYTES)          // 62464

#if __has_builtin(__builtin_amdgcn_global_load_async_to_lds_b128)
#define HAVE_ASYNC 1
#else
#define HAVE_ASYNC 0
#endif

#if __has_builtin(__builtin_amdgcn_s_wait_asynccnt)
#define WAIT_ASYNC(n) __builtin_amdgcn_s_wait_asynccnt(n)
#else
#define WAIT_ASYNC(n) asm volatile("s_wait_asynccnt %0" ::"n"(n) : "memory")
#endif

// pointer casts matching the builtin's signature (AS1 / AS3 v4i pointers)
#define GPTR(p) ((__attribute__((address_space(1))) v4i_t*)(void*)(p))
#define LPTR(p) ((__attribute__((address_space(3))) v4i_t*)(p))

union V16U { v16bf v; v8bf h[2]; };

// ---------------- prep kernels: scatter/gather -> contiguous bf16 K/V^T --

__global__ void init_map_kernel(int* __restrict__ inv) {
  int i = blockIdx.x * blockDim.x + threadIdx.x;
  if (i < NSLOTS) inv[i] = -1;
}

__global__ void fill_map_kernel(const int* __restrict__ slot_mapping,
                                int* __restrict__ inv) {
  int i = blockIdx.x * blockDim.x + threadIdx.x;
  if (i < SEQQ) inv[slot_mapping[i]] = i;
}

// kb layout: [NKVH][LTOT][HD]; vtb layout: [NKVH][HD][LTOT] (pre-transposed)
__global__ void build_kv_kernel(const float* __restrict__ knew,
                                const float* __restrict__ vnew,
                                const float* __restrict__ kcache,
                                const float* __restrict__ vcache,
                                const int* __restrict__ ctx_slots,
                                const int* __restrict__ inv,
                                bf16_t* __restrict__ kb,
                                bf16_t* __restrict__ vtb) {
  size_t i = (size_t)blockIdx.x * blockDim.x + threadIdx.x;
  if (i >= (size_t)LTOT * NKVH * HD) return;
  int d = (int)(i % HD);
  int h = (int)((i / HD) % NKVH);
  int l = (int)(i / ((size_t)HD * NKVH));
  float kv, vv;
  if (l < CTXN) {
    int s = ctx_slots[l];
    int p = inv[s];
    if (p >= 0) {  // context slot overwritten by the new-chunk scatter
      kv = knew[(size_t)p * (NKVH * HD) + (size_t)h * HD + d];
      vv = vnew[(size_t)p * (NKVH * HD) + (size_t)h * HD + d];
    } else {
      kv = kcache[((size_t)s * NKVH + h) * HD + d];
      vv = vcache[((size_t)s * NKVH + h) * HD + d];
    }
  } else {
    int p = l - CTXN;
    kv = knew[(size_t)p * (NKVH * HD) + (size_t)h * HD + d];
    vv = vnew[(size_t)p * (NKVH * HD) + (size_t)h * HD + d];
  }
  kb[((size_t)h * LTOT + l) * HD + d]  = (bf16_t)kv;
  vtb[((size_t)h * HD + d) * LTOT + l] = (bf16_t)vv;
}

// ---------------- flash attention main kernel ---------------------------
// grid = (SEQQ/BM, NKVH); block = 128 (4 waves, one q-head per wave)

__global__ __launch_bounds__(128) void attn_kernel(
    const float* __restrict__ q, const bf16_t* __restrict__ kbuf,
    const bf16_t* __restrict__ vtbuf, float* __restrict__ out) {
  const int qtile = blockIdx.x;
  const int kvh   = blockIdx.y;
  const int q0    = qtile * BM;
  const int tix   = (int)threadIdx.x;
  const int wave  = tix >> 5;
  const int lane  = tix & 31;
  const int head  = kvh * GROUP + wave;
  const int m16   = lane & 15;
  const int half  = lane >> 4;   // 0/1
  const int kb8   = half * 8;    // row/K sub-offset for fragment layouts

  __shared__ __attribute__((aligned(32))) char smem[SMEM_BYTES];
  bf16_t (*ksb)[CH][KPAD] = (bf16_t (*)[CH][KPAD])(smem);           // [2][CH][KPAD]
  bf16_t (*vts)[VPAD]     = (bf16_t (*)[VPAD])(smem + OFF_VT);      // [HD][VPAD]
  bf16_t (*ps)[BM][PPAD]  = (bf16_t (*)[BM][PPAD])(smem + OFF_PS);  // [GROUP][BM][PPAD]
  bf16_t (*qstage)[BM][HD] = (bf16_t (*)[BM][HD])(smem + OFF_VT);   // overlays vts

  // ---- stage Q tile for all 4 heads (scaled, bf16) into vts region ----
  for (int i = tix; i < GROUP * BM * HD; i += 128) {
    int h = i / (BM * HD);
    int r = (i / HD) & (BM - 1);
    int d = i & (HD - 1);
    float val = q[(size_t)(q0 + r) * (NH * HD) + (size_t)(kvh * GROUP + h) * HD + d];
    qstage[h][r][d] = (bf16_t)(val * QSCALE);
  }
  __syncthreads();

  // Q A-fragments: 4 slices of K-dim 32 covering HD=128
  v16bf qa[4];
#pragma unroll
  for (int j = 0; j < 4; ++j) {
    V16U u;
    u.h[0] = *(const v8bf*)&qstage[wave][m16][32 * j + kb8];
    u.h[1] = *(const v8bf*)&qstage[wave][m16][32 * j + kb8 + 16];
    qa[j] = u.v;
  }
  __syncthreads();  // Q region free -> becomes V^T buffer

  v8f acc[8];
#pragma unroll
  for (int sl = 0; sl < 8; ++sl)
#pragma unroll
    for (int e = 0; e < 8; ++e) acc[sl][e] = 0.0f;

  float mrow[8], lrow[8];
#pragma unroll
  for (int r = 0; r < 8; ++r) { mrow[r] = NEGINF; lrow[r] = 0.0f; }

  const int lmax  = CTXN + q0 + BM;       // exclusive causal bound (<= LTOT)
  const int nch   = (lmax + CH - 1) / CH; // keys never exceed LTOT (CH | LTOT)
  int nfull = (CTXN + q0 + 1) / CH;       // chunks needing no masking
  if (nfull > nch) nfull = nch;
  const bf16_t* kh  = kbuf  + (size_t)kvh * LTOT * HD;
  const bf16_t* vth = vtbuf + (size_t)kvh * HD * LTOT;

#if HAVE_ASYNC
  // each wave copies its quarter: 8 x (32 lanes x 16B) per matrix per chunk
  auto issueK = [&](int base, int bsel) {
#pragma unroll
    for (int it = 0; it < 8; ++it) {
      int u  = (wave * 8 + it) * 32 + lane;   // 16B unit index
      int kk = u >> 4;
      int dc = (u & 15) * 8;
      __builtin_amdgcn_global_load_async_to_lds_b128(
          GPTR(kh + (size_t)(base + kk) * HD + dc),
          LPTR(&ksb[bsel][kk][dc]), 0, 0);
    }
  };
  auto issueVT = [&](int base) {
#pragma unroll
    for (int it = 0; it < 8; ++it) {
      int u  = (wave * 8 + it) * 32 + lane;
      int d  = u >> 3;
      int kc = (u & 7) * 8;
      __builtin_amdgcn_global_load_async_to_lds_b128(
          GPTR(vth + (size_t)d * LTOT + base + kc),
          LPTR(&vts[d][kc]), 0, 0);
    }
  };
  issueK(0, 0);  // prime the K pipeline
#endif

  auto chunk_body = [&](int c, auto MASKC) {
    constexpr bool MASKED = decltype(MASKC)::value;
    const int base = c * CH;
    const int bsel = c & 1;
    const bool more = (c + 1 < nch);

#if HAVE_ASYNC
    issueVT(base);                       // vts free (trailing barrier)
    if (more) { issueK(base + CH, bsel ^ 1); WAIT_ASYNC(16); }
    else      { WAIT_ASYNC(8); }         // oldest 8 (this chunk's K) done
    __syncthreads();                     // all waves' K(c) visible
#else
    // sync staging: contiguous 16B copies for K and pre-transposed V
#pragma unroll
    for (int it = 0; it < 8; ++it) {
      int u  = it * 128 + tix;
      int kk = u >> 4;
      int dc = (u & 15) * 8;
      *(uint4*)&ksb[bsel][kk][dc] = *(const uint4*)(kh + (size_t)(base + kk) * HD + dc);
      int d  = u >> 3;
      int kc = (u & 7) * 8;
      *(uint4*)&vts[d][kc] = *(const uint4*)(vth + (size_t)d * LTOT + base + kc);
    }
    __syncthreads();
#endif

    // ---- S = Q * K^T : NF sub-tiles, K-dim 128 = 4 x 32 ----
    v8f s[NF];
#pragma unroll
    for (int nf = 0; nf < NF; ++nf) {
      v16bf bfr[4];
#pragma unroll
      for (int j = 0; j < 4; ++j) {
        V16U u;
        u.h[0] = *(const v8bf*)&ksb[bsel][nf * 16 + m16][32 * j + kb8];
        u.h[1] = *(const v8bf*)&ksb[bsel][nf * 16 + m16][32 * j + kb8 + 16];
        bfr[j] = u.v;
      }
      v8f cacc;
#pragma unroll
      for (int e = 0; e < 8; ++e) cacc[e] = 0.0f;
#pragma unroll
      for (int j = 0; j < 4; ++j)
        cacc = __builtin_amdgcn_wmma_f32_16x16x32_bf16(
            false, qa[j], false, bfr[j], (short)0, cacc, false, false);
      s[nf] = cacc;
    }

    // ---- causal mask (tail chunks only) ----
    if constexpr (MASKED) {
#pragma unroll
      for (int r = 0; r < 8; ++r) {
        int bound = CTXN + q0 + r + kb8;
#pragma unroll
        for (int nf = 0; nf < NF; ++nf) {
          int l = base + nf * 16 + m16;
          if (l > bound) s[nf][r] = NEGINF;
        }
      }
    }

    // ---- per-row max (row = r + kb8, col = lane&15) ----
    float rm[8];
#pragma unroll
    for (int r = 0; r < 8; ++r) {
      float v = s[0][r];
#pragma unroll
      for (int nf = 1; nf < NF; ++nf) v = fmaxf(v, s[nf][r]);
      rm[r] = v;
    }
#pragma unroll
    for (int off = 1; off < 16; off <<= 1)
#pragma unroll
      for (int r = 0; r < 8; ++r)
        rm[r] = fmaxf(rm[r], __shfl_xor(rm[r], off, 32));

    // ---- online softmax update (exp2 domain) ----
    float rs[8];
#pragma unroll
    for (int r = 0; r < 8; ++r) {
      float mnew = fmaxf(mrow[r], rm[r]);
      float sc   = __builtin_amdgcn_exp2f(mrow[r] - mnew);
      mrow[r] = mnew;
      lrow[r] *= sc;
#pragma unroll
      for (int sl = 0; sl < 8; ++sl) acc[sl][r] *= sc;
      float rsum = 0.0f;
#pragma unroll
      for (int nf = 0; nf < NF; ++nf) {
        float p = __builtin_amdgcn_exp2f(s[nf][r] - mnew);
        rsum += p;
        ps[wave][r + kb8][nf * 16 + m16] = (bf16_t)p;
      }
      rs[r] = rsum;
    }
#pragma unroll
    for (int off = 1; off < 16; off <<= 1)
#pragma unroll
      for (int r = 0; r < 8; ++r)
        rs[r] += __shfl_xor(rs[r], off, 32);
#pragma unroll
    for (int r = 0; r < 8; ++r) lrow[r] += rs[r];

#if HAVE_ASYNC
    if (more) WAIT_ASYNC(8); else WAIT_ASYNC(0);  // this chunk's V^T done
    __syncthreads();                              // all waves' V^T visible
#endif

    // ---- O += P * V : two K=32 P fragments x 8 hd sub-tiles ----
#pragma unroll
    for (int t2 = 0; t2 < 2; ++t2) {
      V16U pu;
      pu.h[0] = *(const v8bf*)&ps[wave][m16][32 * t2 + kb8];
      pu.h[1] = *(const v8bf*)&ps[wave][m16][32 * t2 + kb8 + 16];
      v16bf pa = pu.v;
#pragma unroll
      for (int sl = 0; sl < 8; ++sl) {
        V16U u;
        u.h[0] = *(const v8bf*)&vts[sl * 16 + m16][32 * t2 + kb8];
        u.h[1] = *(const v8bf*)&vts[sl * 16 + m16][32 * t2 + kb8 + 16];
        acc[sl] = __builtin_amdgcn_wmma_f32_16x16x32_bf16(
            false, pa, false, u.v, (short)0, acc[sl], false, false);
      }
    }
    __syncthreads();   // release vts / K buffer for the next chunk
  };

  int c = 0;
  for (; c < nfull; ++c) chunk_body(c, std::false_type{});   // fast path
  for (; c < nch; ++c)   chunk_body(c, std::true_type{});    // masked tail

  // ---- epilogue: O / l, fp32 out ----
#pragma unroll
  for (int r = 0; r < 8; ++r) {
    float inv = 1.0f / lrow[r];
    size_t row = (size_t)(q0 + r + kb8) * (NH * HD) + (size_t)head * HD;
#pragma unroll
    for (int sl = 0; sl < 8; ++sl)
      out[row + sl * 16 + m16] = acc[sl][r] * inv;
  }
}

// ---------------- launch -------------------------------------------------

extern "C" void kernel_launch(void* const* d_in, const int* in_sizes, int n_in,
                              void* d_out, int out_size, void* d_ws, size_t ws_size,
                              hipStream_t stream) {
  const float* q  = (const float*)d_in[0];
  const float* k  = (const float*)d_in[1];
  const float* v  = (const float*)d_in[2];
  const float* kc = (const float*)d_in[3];
  const float* vc = (const float*)d_in[4];
  const int* slot_mapping  = (const int*)d_in[5];
  const int* context_slots = (const int*)d_in[6];
  float* out = (float*)d_out;

  char* ws = (char*)d_ws;
  int* inv = (int*)ws;
  bf16_t* kb  = (bf16_t*)(ws + (size_t)NSLOTS * sizeof(int));
  bf16_t* vtb = kb + (size_t)NKVH * LTOT * HD;

  init_map_kernel<<<(NSLOTS + 255) / 256, 256, 0, stream>>>(inv);
  fill_map_kernel<<<(SEQQ + 255) / 256, 256, 0, stream>>>(slot_mapping, inv);

  size_t kv_elems = (size_t)LTOT * NKVH * HD;
  build_kv_kernel<<<(unsigned)((kv_elems + 255) / 256), 256, 0, stream>>>(
      k, v, kc, vc, context_slots, inv, kb, vtb);

  dim3 grid(SEQQ / BM, NKVH);
  attn_kernel<<<grid, 128, 0, stream>>>(q, kb, vtb, out);
}
